// Model_4483945857105
// MI455X (gfx1250) — matrix operand order repaired and checked by reading, not compiled, not run on hardware
//
#include <hip/hip_runtime.h>
#include <hip/hip_bf16.h>

// ---------------------------------------------------------------------------
// LSTM (B=64, L=512, H=512, D=21) + FC [B*L,512]x[512,2016] for MI455X gfx1250.
//  - prep kernel: fp32->bf16 weights, K-padded Wx/x (21->32), fused bias,
//    zeroed barrier counters (deterministic graph replay).
//  - lstm kernel: 64 WGs = 4 batch-groups(16 rows = M-tile) x 16 h-slices.
//    512 thr/WG, 16 waves; each 16x16 gate tile owned by a wave pair that
//    splits K=512 in half -> 8 hoisted Wh B-fragments (64 VGPRs, register
//    resident, no scratch spills). A-fragment loads are pipelined via
//    sched_group_barrier so each WMMA waits dscnt<=2 (LDS latency hidden
//    behind the accumulator-chained WMMAs). Branch-free gate nonlinearities.
//  - fc kernel: one 16x16 tile per wave, K=512 -> 16 WMMAs, VMEM-read groups
//    pipelined; 264MB output is the HBM-bound term (~11us @ 23.3 TB/s).
// ---------------------------------------------------------------------------

typedef __attribute__((ext_vector_type(16))) __bf16 v16bf;
typedef __attribute__((ext_vector_type(8)))  __bf16 v8bf;
typedef __attribute__((ext_vector_type(8)))  float  v8f;

union Frag16 { v16bf v; v8bf h[2]; };

#if __has_builtin(__builtin_amdgcn_sched_group_barrier)
#define SG(mask, n) __builtin_amdgcn_sched_group_barrier((mask), (n), 0)
#else
#define SG(mask, n)
#endif
// sched_group_barrier masks: 0x8 = MFMA/WMMA, 0x20 = VMEM read, 0x100 = DS read

#define B_   64
#define L_   512
#define D_   21
#define H_   512
#define G4_  2048   /* 4*H  */
#define NFC_ 2016   /* PRED_LEN*ENC_IN */

#define LOG2E_ 1.44269504088896340736f

// workspace byte offsets (all 32B aligned)
#define WH_OFF    0u         /* Wh  bf16 [2048][512]  : 2,097,152 B */
#define WFC_OFF   2097152u   /* Wfc bf16 [2016][512]  : 2,064,384 B */
#define WXP_OFF   4161536u   /* Wx  bf16 [2048][32]   :   131,072 B (K-padded) */
#define BXH_OFF   4292608u   /* bx+bh f32 [2048]      :     8,192 B */
#define XPAD_OFF  4300800u   /* x bf16 [B*L][32]      : 2,097,152 B (K-padded) */
#define CNT_OFF   6397952u   /* 4 barrier counters    :       256 B */
#define HALL_OFF  6398208u   /* h_all bf16 [B*L][512] :33,554,432 B */

// branch-free transcendentals: v_exp_f32 + v_rcp_f32, no libm range reduction
__device__ __forceinline__ float fast_sigmoid(float x) {
  return __builtin_amdgcn_rcpf(1.0f + __builtin_amdgcn_exp2f(-LOG2E_ * x));
}
__device__ __forceinline__ float fast_tanh(float x) {
  return 1.0f -
         2.0f * __builtin_amdgcn_rcpf(
                    1.0f + __builtin_amdgcn_exp2f(2.0f * LOG2E_ * x));
}

// ---------------------------------------------------------------------------
__global__ void prep_kernel(const float* __restrict__ x,
                            const float* __restrict__ Wx,
                            const float* __restrict__ bx,
                            const float* __restrict__ Wh,
                            const float* __restrict__ bh,
                            const float* __restrict__ Wfc,
                            unsigned char* __restrict__ ws) {
  __bf16*       whb  = (__bf16*)(ws + WH_OFF);
  __bf16*       wfcb = (__bf16*)(ws + WFC_OFF);
  __bf16*       wxp  = (__bf16*)(ws + WXP_OFF);
  float*        bxh  = (float*)(ws + BXH_OFF);
  __bf16*       xb   = (__bf16*)(ws + XPAD_OFF);
  unsigned int* cnt  = (unsigned int*)(ws + CNT_OFF);

  long i = (long)blockIdx.x * blockDim.x + threadIdx.x;
  long n;
  n = (long)G4_ * H_;                         // Wh
  if (i < n) { whb[i] = (__bf16)Wh[i]; return; }
  i -= n;
  n = (long)NFC_ * H_;                        // Wfc
  if (i < n) { wfcb[i] = (__bf16)Wfc[i]; return; }
  i -= n;
  n = (long)G4_ * 32;                         // Wx padded K 21->32
  if (i < n) {
    long g = i >> 5; int d = (int)(i & 31);
    wxp[i] = (__bf16)(d < D_ ? Wx[g * D_ + d] : 0.0f);
    return;
  }
  i -= n;
  n = G4_;                                    // fused bias
  if (i < n) { bxh[i] = bx[i] + bh[i]; return; }
  i -= n;
  n = (long)B_ * L_ * 32;                     // x padded K 21->32
  if (i < n) {
    long r = i >> 5; int d = (int)(i & 31);
    xb[i] = (__bf16)(d < D_ ? x[r * D_ + d] : 0.0f);
    return;
  }
  i -= n;
  if (i < 4) cnt[i] = 0u;                     // barrier counters (every launch)
}

// ---------------------------------------------------------------------------
// 64 WGs x 512 threads (16 waves). blockIdx = bg*16 + hs.
__global__ void __launch_bounds__(512, 4)
lstm_kernel(unsigned char* __restrict__ ws) {
  const __bf16* whb = (const __bf16*)(ws + WH_OFF);
  const __bf16* wxp = (const __bf16*)(ws + WXP_OFF);
  const float*  bxh = (const float*)(ws + BXH_OFF);
  const __bf16* xb  = (const __bf16*)(ws + XPAD_OFF);
  __bf16*       hall = (__bf16*)(ws + HALL_OFF);
  unsigned int* cnt  = (unsigned int*)(ws + CNT_OFF);

  __shared__ __bf16 hsh[16 * H_];        // h_{t-1} stage: 16 x 512 bf16, 16KB
  __shared__ float  gsh[16][16][16];     // per-wave partial gate tiles, 16KB

  const int tid   = threadIdx.x;
  const int wave  = tid >> 5;            // 0..15
  const int lane  = tid & 31;
  const int lm    = lane & 15;           // M (A) or N (B/C) index inside tile
  const int lh    = lane >> 4;           // half-wave selector
  const int bg    = blockIdx.x >> 4;     // batch group 0..3
  const int hs    = blockIdx.x & 15;     // h-slice   0..15
  const int tl    = wave >> 1;           // gate tile 0..7
  const int khalf = wave & 1;            // K half: 0 -> K[0,256), 1 -> K[256,512)
  const int gb    = tl >> 1;             // 0=i 1=f 2=g 3=o
  const int sub   = tl & 1;
  const int n0    = gb * H_ + hs * 32 + sub * 16;  // gate column base

  // ---- loop-invariant operands hoisted to VGPRs (8 frags = 64 VGPRs) ----
  v16bf whf[8];
#pragma unroll
  for (int kk = 0; kk < 8; ++kk) {
    const int kabs = khalf * 8 + kk;
    whf[kk] = *(const v16bf*)(whb + (long)(n0 + lm) * H_ + kabs * 32 + lh * 16);
  }
  const v16bf wxf  = *(const v16bf*)(wxp + (long)(n0 + lm) * 32 + lh * 16);
  const float bias = bxh[n0 + lm];

  // this wave's A-fragment base inside the LDS h tile
  const __bf16* const abase = hsh + lm * H_ + khalf * 256 + lh * 16;

  float creg = 0.0f;                     // this thread's cell state (1 entry)

  for (int t = 0; t < L_; ++t) {
    // ---- stage h_{t-1}[16][512] bf16 into LDS (32B per thread) ----------
    {
      const int m = tid >> 5, part = tid & 31;       // 32 threads per row
      uint4* dst = (uint4*)(hsh + m * H_ + part * 16);
      if (t == 0) {
        uint4 z = {0u, 0u, 0u, 0u};
        dst[0] = z; dst[1] = z;
      } else {
        const uint4* src = (const uint4*)(hall +
            ((long)(bg * 16 + m) * L_ + (t - 1)) * H_ + part * 16);
        dst[0] = src[0]; dst[1] = src[1];
      }
    }
    __syncthreads();

    // ---- partial gates tile ---------------------------------------------
    v8f acc;
    if (khalf == 0) {
#pragma unroll
      for (int r = 0; r < 8; ++r) acc[r] = bias;     // bias seed
    } else {
#pragma unroll
      for (int r = 0; r < 8; ++r) acc[r] = 0.0f;
      Frag16 a;                                      // x-projection WMMA
      const __bf16* xrow = xb + ((long)(bg * 16 + lm) * L_ + t) * 32 + lh * 16;
      a.h[0] = *(const v8bf*)(xrow);
      a.h[1] = *(const v8bf*)(xrow + 8);
      acc = __builtin_amdgcn_wmma_f32_16x16x32_bf16(
          false, a.v, false, wxf, (short)0, acc, false, false);
    }

    // Ping-pong pipelined K loop; scheduling pinned below so each WMMA
    // waits only dscnt<=2 while the next fragment's loads are in flight.
    Frag16 a0, a1;
    a0.h[0] = *(const v8bf*)(abase);
    a0.h[1] = *(const v8bf*)(abase + 8);
    a1.h[0] = *(const v8bf*)(abase + 32);
    a1.h[1] = *(const v8bf*)(abase + 40);
#pragma unroll
    for (int kk = 0; kk < 8; kk += 2) {
      acc = __builtin_amdgcn_wmma_f32_16x16x32_bf16(
          false, a0.v, false, whf[kk], (short)0, acc, false, false);
      if (kk + 2 < 8) {
        const __bf16* p = abase + (kk + 2) * 32;
        a0.h[0] = *(const v8bf*)(p);
        a0.h[1] = *(const v8bf*)(p + 8);
      }
      acc = __builtin_amdgcn_wmma_f32_16x16x32_bf16(
          false, a1.v, false, whf[kk + 1], (short)0, acc, false, false);
      if (kk + 3 < 8) {
        const __bf16* p = abase + (kk + 3) * 32;
        a1.h[0] = *(const v8bf*)(p);
        a1.h[1] = *(const v8bf*)(p + 8);
      }
    }
    // Pin the pipeline: DS(4) [WMMA DS(2)]x6 WMMA WMMA
    SG(0x100, 4);
#pragma unroll
    for (int i = 0; i < 6; ++i) { SG(0x8, 1); SG(0x100, 2); }
    SG(0x8, 1);
    SG(0x8, 1);

    // ---- spill partial tiles (C layout: M = r + 8*lh, N = lm) -----------
#pragma unroll
    for (int r = 0; r < 8; ++r)
      gsh[wave][r + 8 * lh][lm] = acc[r];
    __syncthreads();

    // ---- elementwise LSTM cell: 512 entries / 512 threads ---------------
    {
      const int m = tid >> 5, d = tid & 31;
      const int s = d >> 4, col = d & 15;
      const int ti = (0 + s) * 2;                    // i tile's wave pair
      const int tf = (2 + s) * 2;
      const int tg = (4 + s) * 2;
      const int to = (6 + s) * 2;
      const float iv = gsh[ti][m][col] + gsh[ti + 1][m][col];
      const float fv = gsh[tf][m][col] + gsh[tf + 1][m][col];
      const float gv = gsh[tg][m][col] + gsh[tg + 1][m][col];
      const float ov = gsh[to][m][col] + gsh[to + 1][m][col];
      creg = fast_sigmoid(fv) * creg + fast_sigmoid(iv) * fast_tanh(gv);
      const float hv = fast_sigmoid(ov) * fast_tanh(creg);
      hall[((long)(bg * 16 + m) * L_ + t) * H_ + hs * 32 + d] = (__bf16)hv;
    }

    // ---- inter-WG barrier among the 16 h-slice WGs of this group --------
    if (t < L_ - 1) {
      __threadfence();        // this thread's h stores -> agent visible
      __syncthreads();        // all threads fenced before signaling
      if (tid == 0)
        __hip_atomic_fetch_add(&cnt[bg], 1u, __ATOMIC_RELEASE,
                               __HIP_MEMORY_SCOPE_AGENT);
      const unsigned int target = 16u * (unsigned int)(t + 1);
      while (__hip_atomic_load(&cnt[bg], __ATOMIC_ACQUIRE,
                               __HIP_MEMORY_SCOPE_AGENT) < target)
        __builtin_amdgcn_s_sleep(2);
      __syncthreads();
    }
  }
}

// ---------------------------------------------------------------------------
// FC: out[32768,2016] = h_all(bf16) @ Wfc^T + bfc.  One 16x16 tile per wave.
// grid = 258048 tiles / 8 waves = 32256 blocks.
__global__ void __launch_bounds__(256)
fc_kernel(const float* __restrict__ bfc,
          const unsigned char* __restrict__ ws,
          float* __restrict__ out) {
  const __bf16* hall = (const __bf16*)(ws + HALL_OFF);
  const __bf16* wfcb = (const __bf16*)(ws + WFC_OFF);

  const int wave = threadIdx.x >> 5;
  const int lane = threadIdx.x & 31;
  const int lm   = lane & 15;
  const int lh   = lane >> 4;

  const int tile  = blockIdx.x * 8 + wave;   // 0..258047
  const int mtile = tile & 2047;             // M tiles: 32768/16
  const int ntile = tile >> 11;              // N tiles: 2016/16 = 126

  const __bf16* ap = hall + (long)(mtile * 16 + lm) * H_ + lh * 16;
  const __bf16* bp = wfcb + (long)(ntile * 16 + lm) * H_ + lh * 16;

  __builtin_prefetch(bp, 0, 1);              // global_prefetch_b8 into GL2

  // ping-pong pipelined K loop (schedule pinned below)
  Frag16 a0, a1;
  v16bf  b0, b1;
  a0.h[0] = *(const v8bf*)(ap);
  a0.h[1] = *(const v8bf*)(ap + 8);
  b0      = *(const v16bf*)(bp);
  a1.h[0] = *(const v8bf*)(ap + 32);
  a1.h[1] = *(const v8bf*)(ap + 40);
  b1      = *(const v16bf*)(bp + 32);

  v8f acc = {};
#pragma unroll
  for (int kk = 0; kk < 16; kk += 2) {
    acc = __builtin_amdgcn_wmma_f32_16x16x32_bf16(
        false, a0.v, false, b0, (short)0, acc, false, false);
    if (kk + 2 < 16) {
      const __bf16* an = ap + (kk + 2) * 32;
      a0.h[0] = *(const v8bf*)(an);
      a0.h[1] = *(const v8bf*)(an + 8);
      b0      = *(const v16bf*)(bp + (kk + 2) * 32);
    }
    acc = __builtin_amdgcn_wmma_f32_16x16x32_bf16(
        false, a1.v, false, b1, (short)0, acc, false, false);
    if (kk + 3 < 16) {
      const __bf16* an = ap + (kk + 3) * 32;
      a1.h[0] = *(const v8bf*)(an);
      a1.h[1] = *(const v8bf*)(an + 8);
      b1      = *(const v16bf*)(bp + (kk + 3) * 32);
    }
  }
  // Pin the pipeline: VM(8) [WMMA VM(4)]x14 WMMA WMMA
  SG(0x20, 8);
#pragma unroll
  for (int i = 0; i < 14; ++i) { SG(0x8, 1); SG(0x20, 4); }
  SG(0x8, 1);
  SG(0x8, 1);

  const int   n    = ntile * 16 + lm;
  const float bias = bfc[n];
#pragma unroll
  for (int r = 0; r < 8; ++r) {
    const int m = mtile * 16 + r + 8 * lh;   // C layout: M = r + 8*(lane/16)
    out[(long)m * NFC_ + n] = acc[r] + bias;
  }
}

// ---------------------------------------------------------------------------
extern "C" void kernel_launch(void* const* d_in, const int* in_sizes, int n_in,
                              void* d_out, int out_size, void* d_ws, size_t ws_size,
                              hipStream_t stream) {
  (void)in_sizes; (void)n_in; (void)out_size; (void)ws_size;
  const float* x   = (const float*)d_in[0];
  const float* Wx  = (const float*)d_in[1];
  const float* bx  = (const float*)d_in[2];
  const float* Wh  = (const float*)d_in[3];
  const float* bh  = (const float*)d_in[4];
  const float* Wfc = (const float*)d_in[5];
  const float* bfc = (const float*)d_in[6];
  unsigned char* ws = (unsigned char*)d_ws;
  float* out = (float*)d_out;

  const long NC = 1048576L + 1032192L + 65536L + 2048L + 1048576L + 4L;
  const int  cb = (int)((NC + 255) / 256);

  prep_kernel<<<cb, 256, 0, stream>>>(x, Wx, bx, Wh, bh, Wfc, ws);
  lstm_kernel<<<64, 512, 0, stream>>>(ws);
  fc_kernel<<<32256, 256, 0, stream>>>(bfc, ws, out);
}
